// WaitKAttentionLayer_15496242004632
// MI455X (gfx1250) — matrix operand order repaired and checked by reading, not compiled, not run on hardware
//
#include <hip/hip_runtime.h>
#include <hip/hip_bf16.h>

// ---------------------------------------------------------------------------
// WaitK attention for MI455X (gfx1250, wave32, WMMA bf16 + TDM path)
// S=2048, B=32, D=E=H=1024, ptr = 9
// ---------------------------------------------------------------------------

#define S_LEN 2048
#define BSZ   32
#define HID   1024
#define PTR   9
#define NEG_INF_F (-1e9f)

#define APITCH 1028   // f32 elements per LDS row: 16B-aligned frags, 2-cycle banks

typedef __bf16 v16bf __attribute__((ext_vector_type(16)));
typedef float  v8f   __attribute__((ext_vector_type(8)));
typedef unsigned int u32x4 __attribute__((ext_vector_type(4)));
typedef int          i32x4 __attribute__((ext_vector_type(4)));
typedef int          i32x8 __attribute__((ext_vector_type(8)));

union FragA { v16bf v; unsigned int d[8]; };
union FragB { v16bf v; uint4 q[2]; };

__device__ __forceinline__ unsigned int pk_trunc(float lo, float hi) {
    // {bf16(hi), bf16(lo)} via one v_perm_b32 (truncating round)
    return __builtin_amdgcn_perm(__float_as_uint(hi), __float_as_uint(lo), 0x07060302u);
}

__device__ __forceinline__ unsigned short bf16of(float a) {
    unsigned int ua = __float_as_uint(a);
    ua += 0x7FFFu + ((ua >> 16) & 1u);
    return (unsigned short)(ua >> 16);
}

// ---------------------------------------------------------------------------
// Swizzle W_enc (1024x1024 f32, k-major) into bf16 B-fragment order:
// block (nt*32+ks) is 1KB; lane L owns bytes [L*32, L*32+32):
//   N = nt*16 + (L&15),  K = ks*32 + (L>=16?8:0) + (j&7) + (j>=8?16:0), j=0..15
// ---------------------------------------------------------------------------
__global__ __launch_bounds__(256) void wswz_kernel(const float* __restrict__ We,
                                                   unsigned short* __restrict__ Wz) {
    int blk = blockIdx.x;            // 0..2047  (nt = blk>>5, ks = blk&31)
    int nt  = blk >> 5;
    int ks  = blk & 31;
    for (int t = threadIdx.x; t < 512; t += 256) {
        int lane = t >> 4;
        int j    = t & 15;
        int n = nt * 16 + (lane & 15);
        int k = ks * 32 + ((lane >> 4) << 3) + (j & 7) + ((j >> 3) << 4);
        Wz[(size_t)blk * 512 + lane * 16 + j] = bf16of(We[(size_t)k * HID + n]);
    }
}

// ---------------------------------------------------------------------------
// dec = input @ W_dec + b   (32 x 1024)
// ---------------------------------------------------------------------------
__global__ __launch_bounds__(256) void dec_kernel(const float* __restrict__ inp,
                                                  const float* __restrict__ Wd,
                                                  const float* __restrict__ bias,
                                                  float* __restrict__ dec) {
    int b = blockIdx.x;                              // 32
    int h = blockIdx.y * 256 + threadIdx.x;          // 4 * 256
    float acc = bias[h];
    const float* irow = inp + (size_t)b * HID;
    for (int d = 0; d < HID; ++d)
        acc = fmaf(irow[d], Wd[(size_t)d * HID + h], acc);
    dec[(size_t)b * HID + h] = acc;
}

// ---------------------------------------------------------------------------
// Energy: e[s,b] = v . tanh(src[s,b,:] @ W_enc + dec[b,:]),  mask -> -1e9
// One block = one 16-row tile (fixed s, b0 = 16*(tile&1)); 8 waves.
// A tile staged raw f32 into LDS by the Tensor Data Mover (iterate mode,
// 16 rows, lds pitch 1028 dwords); f32->bf16 at fragment assembly (v_perm).
// ---------------------------------------------------------------------------
__global__ __launch_bounds__(256) void energy_kernel(const float* __restrict__ src,
                                                     const unsigned char* __restrict__ mask,
                                                     const float* __restrict__ dec,
                                                     const unsigned short* __restrict__ Wz,
                                                     const float* __restrict__ vvec,
                                                     float* __restrict__ e_out) {
    __shared__ float ashf[16 * APITCH];   // ~64.3 KB raw f32 A tile
    __shared__ float esh[16];

    const int tid  = threadIdx.x;
    const int tile = blockIdx.x;          // 0..4095
    const int s    = tile >> 1;
    const int b0   = (tile & 1) << 4;

    if (tid < 16) esh[tid] = 0.0f;

    const int wv = tid >> 5;      // wave 0..7
    const int ln = tid & 31;      // lane
    const int lm = ln & 15;       // M row in A-frag / N col in C
    const int hi = ln >> 4;       // lane half

#if __has_builtin(__builtin_amdgcn_tensor_load_to_lds)
    // ---- TDM staging: 16 iterations of one 1024-dword row ----
    if (wv == 0) {
        unsigned long long ga =
            (unsigned long long)(const void*)(src + (size_t)(s * BSZ + b0) * HID);
        unsigned int ldsb = (unsigned int)(size_t)(void*)ashf;  // LDS byte offset

        u32x4 g0 = { 1u,                                   // count=1, user mode
                     ldsb,                                 // lds_addr
                     (unsigned int)(ga & 0xFFFFFFFFu),     // global_addr[31:0]
                     (unsigned int)((ga >> 32) & 0x1FFFFFFu) | (2u << 30) }; // type=2
        i32x8 g1 = { (2 << 16) | (1 << 19),   // data_size=4B, iterate_enable
                     (int)(1024u << 16),      // tensor_dim0 = 1024 (lo16 @ [31:16])
                     0,                       // tensor_dim0 hi / tensor_dim1 lo
                     (int)(1u | (1024u << 16)), // tensor_dim1 = 65536; tile_dim0 = 1024
                     1,                       // tile_dim1 = 1, tile_dim2 = 0
                     1024,                    // tensor_dim0_stride = 1024
                     0, 0 };
        i32x4 g2 = { 0,                       // tensor_dim2
                     APITCH,                  // lds_addr_increment (1028 dwords)
                     1024,                    // global_addr_increment
                     (int)(15u << 16) };      // iterate_count = 15 (16 rows)
        i32x4 g3 = { 0, 0, 0, 0 };
#if __has_include(<hip/amd_detail/amd_gfx1250_TDM.h>)
        i32x8 g4 = { 0, 0, 0, 0, 0, 0, 0, 0 };
        __builtin_amdgcn_tensor_load_to_lds(g0, g1, g2, g3, g4, 0);
#else
        __builtin_amdgcn_tensor_load_to_lds(g0, g1, g2, g3, 0);
#endif
        __builtin_amdgcn_s_wait_tensorcnt(0);
    }
#else
    // ---- fallback: manual staging of the same f32 layout ----
    {
        int r     = tid >> 4;
        int cbase = (tid & 15) << 6;
        const float4* gp = (const float4*)(src + ((size_t)(s * BSZ + b0 + r)) * HID + cbase);
        float* lrow = &ashf[(size_t)r * APITCH + cbase];
#pragma unroll
        for (int i = 0; i < 16; ++i)
            *(float4*)(lrow + i * 4) = gp[i];
    }
#endif
    __syncthreads();

    // ---- WMMA main loop ----
    const uint4* wzq = (const uint4*)Wz;

    v8f zero = {0.f, 0.f, 0.f, 0.f, 0.f, 0.f, 0.f, 0.f};
    v8f c[8];
#pragma unroll
    for (int i = 0; i < 8; ++i) c[i] = zero;

    const float* arow = &ashf[(size_t)lm * APITCH];

    for (int ks = 0; ks < 32; ++ks) {
        int col = (ks << 5) + (hi << 3);
        float4 f0 = *(const float4*)(arow + col);
        float4 f1 = *(const float4*)(arow + col + 4);
        float4 f2 = *(const float4*)(arow + col + 16);
        float4 f3 = *(const float4*)(arow + col + 20);

        FragA a;
        a.d[0] = pk_trunc(f0.x, f0.y);
        a.d[1] = pk_trunc(f0.z, f0.w);
        a.d[2] = pk_trunc(f1.x, f1.y);
        a.d[3] = pk_trunc(f1.z, f1.w);
        a.d[4] = pk_trunc(f2.x, f2.y);
        a.d[5] = pk_trunc(f2.z, f2.w);
        a.d[6] = pk_trunc(f3.x, f3.y);
        a.d[7] = pk_trunc(f3.z, f3.w);

        // prefetch next K-step's B block for this wave's first N-tile
        __builtin_prefetch((const void*)(wzq + ((size_t)(wv * 32 + ks + 1)) * 64 + ln * 2), 0, 3);

#pragma unroll
        for (int i = 0; i < 8; ++i) {
            int nt = wv + (i << 3);
            const uint4* bp = wzq + ((size_t)(nt * 32 + ks)) * 64 + ln * 2;
            FragB bb;
            bb.q[0] = bp[0];
            bb.q[1] = bp[1];
            c[i] = __builtin_amdgcn_wmma_f32_16x16x32_bf16(
                false, a.v, false, bb.v, (short)0, c[i], false, false);
        }
    }

    // ---- fused epilogue: tanh(+dec) . v, reduce over h ----
    float eacc[8];
#pragma unroll
    for (int j = 0; j < 8; ++j) eacc[j] = 0.0f;

#pragma unroll
    for (int i = 0; i < 8; ++i) {
        int nt = wv + (i << 3);
        int h  = (nt << 4) + lm;
        float vh = vvec[h];
#pragma unroll
        for (int j = 0; j < 8; ++j) {
            int brow = b0 + (hi << 3) + j;
            float dv = dec[(size_t)brow * HID + h];
            eacc[j] += tanhf(c[i][j] + dv) * vh;
        }
    }

    // reduce across the 16 lanes of each half (h dimension)
#pragma unroll
    for (int j = 0; j < 8; ++j) {
        float sum = eacc[j];
        sum += __shfl_xor(sum, 1, 32);
        sum += __shfl_xor(sum, 2, 32);
        sum += __shfl_xor(sum, 4, 32);
        sum += __shfl_xor(sum, 8, 32);
        if (lm == 0) atomicAdd(&esh[(hi << 3) + j], sum);
    }
    __syncthreads();

    if (tid < 16) {
        int b = b0 + tid;
        float ev = esh[tid];
        if (mask[(size_t)s * BSZ + b]) ev = NEG_INF_F;
        e_out[(size_t)s * BSZ + b] = ev;
    }
}

// ---------------------------------------------------------------------------
// Finalize: global max over s, softmax over s<PTR with eps, weighted context
// ---------------------------------------------------------------------------
__global__ __launch_bounds__(256) void finalize_kernel(const float* __restrict__ e,
                                                       const float* __restrict__ src,
                                                       float* __restrict__ out) {
    __shared__ float red[256];
    __shared__ float beta[16];
    const int b   = blockIdx.x;
    const int tid = threadIdx.x;

    float m = -3.4e38f;
    for (int s = tid; s < S_LEN; s += 256)
        m = fmaxf(m, e[(size_t)s * BSZ + b]);
    red[tid] = m;
    __syncthreads();
    for (int st = 128; st > 0; st >>= 1) {
        if (tid < st) red[tid] = fmaxf(red[tid], red[tid + st]);
        __syncthreads();
    }
    if (tid == 0) {
        float M = red[0];
        float tmp[PTR];
        float sum = 0.0f;
        for (int s = 0; s < PTR; ++s) {
            tmp[s] = __expf(e[(size_t)s * BSZ + b] - M) + 1e-8f;
            sum += tmp[s];
        }
        float inv = 1.0f / sum;
        for (int s = 0; s < PTR; ++s) beta[s] = tmp[s] * inv;
    }
    __syncthreads();

    for (int h = tid; h < HID; h += 256) {
        float acc = 0.0f;
        for (int s = 0; s < PTR; ++s)
            acc = fmaf(beta[s], src[((size_t)s * BSZ + b) * HID + h], acc);
        out[(size_t)b * HID + h] = acc;
    }
}

// ---------------------------------------------------------------------------
// alpha: one-hot at s = PTR-1 = 8   (S,B)
// ---------------------------------------------------------------------------
__global__ __launch_bounds__(256) void alpha_kernel(float* __restrict__ alpha) {
    int idx = blockIdx.x * 256 + threadIdx.x;     // 0..65535
    alpha[idx] = ((idx >> 5) == (PTR - 1)) ? 1.0f : 0.0f;
}

// ---------------------------------------------------------------------------
extern "C" void kernel_launch(void* const* d_in, const int* in_sizes, int n_in,
                              void* d_out, int out_size, void* d_ws, size_t ws_size,
                              hipStream_t stream) {
    const float*         input = (const float*)d_in[0];         // (32,1024)
    const float*         src   = (const float*)d_in[1];         // (2048,32,1024)
    const unsigned char* mask  = (const unsigned char*)d_in[2]; // (2048,32) bool
    const float*         Wd    = (const float*)d_in[3];         // (1024,1024)
    const float*         We    = (const float*)d_in[4];         // (1024,1024)
    const float*         bias  = (const float*)d_in[5];         // (1024,)
    const float*         v     = (const float*)d_in[6];         // (1024,)

    float* out   = (float*)d_out;              // [32*1024 output | 2048*32 alpha]
    float* alpha = out + BSZ * HID;

    char* ws = (char*)d_ws;
    unsigned short* Wz  = (unsigned short*)ws;                        // 2 MB
    float*          dec = (float*)(ws + (2u << 20));                  // 128 KB
    float*          e   = (float*)(ws + (2u << 20) + (128u << 10));   // 256 KB

    wswz_kernel<<<2048, 256, 0, stream>>>(We, Wz);
    dec_kernel<<<dim3(32, 4), 256, 0, stream>>>(input, Wd, bias, dec);
    energy_kernel<<<4096, 256, 0, stream>>>(src, mask, dec, Wz, v, e);
    finalize_kernel<<<32, 256, 0, stream>>>(e, src, out);
    alpha_kernel<<<256, 256, 0, stream>>>(alpha);
}
